// FlashMultiheadAttention_51058571215191
// MI455X (gfx1250) — compile-verified, matrix operand order checked by reading, not run on hardware
//
#include <hip/hip_runtime.h>
#include <hip/hip_bf16.h>

#define E_DIM 1280
#define HEADS 20
#define HDIM  64

typedef __attribute__((ext_vector_type(16))) __bf16 v16bf;
typedef __attribute__((ext_vector_type(8)))  __bf16 v8bf;
typedef __attribute__((ext_vector_type(8)))  float  v8f;

__device__ __forceinline__ v8f wmma_bf16(v16bf a, v16bf b, v8f c) {
    // (neg_a, A, neg_b, B, c_mod, C, reuse_a, reuse_b)
    return __builtin_amdgcn_wmma_f32_16x16x32_bf16(false, a, false, b, (short)0, c,
                                                   false, false);
}

// Load a 16x32 bf16 A/B tile fragment for this lane.
// Layout (ISA 7.12.2, 16-bit A 16x32): lanes 0-15 hold K=0..7,16..23 of row M=lane,
// lanes 16-31 hold K=8..15,24..31. Per-lane: two contiguous 16B loads.
__device__ __forceinline__ v16bf load_tile16(const __bf16* base, int stride,
                                             int row, int kbase, int hi) {
    const __bf16* p0 = base + (size_t)row * stride + kbase + hi * 8;
    const __bf16* p1 = p0 + 16;
    v8bf lo = *(const v8bf*)p0;
    v8bf hi8 = *(const v8bf*)p1;
    v16bf r;
#pragma unroll
    for (int j = 0; j < 8; ++j) { r[j] = lo[j]; r[j + 8] = hi8[j]; }
    return r;
}

__device__ __forceinline__ float half_red_max(float v) {
    v = fmaxf(v, __shfl_xor(v, 1));
    v = fmaxf(v, __shfl_xor(v, 2));
    v = fmaxf(v, __shfl_xor(v, 4));
    v = fmaxf(v, __shfl_xor(v, 8));
    return v;
}
__device__ __forceinline__ float half_red_sum(float v) {
    v += __shfl_xor(v, 1);
    v += __shfl_xor(v, 2);
    v += __shfl_xor(v, 4);
    v += __shfl_xor(v, 8);
    return v;
}

// Fragment set for one 32-wide K chunk of the GEMM: two A tiles + four B tiles.
__device__ __forceinline__ void load_frag_set(const __bf16* __restrict__ amat,
                                              const __bf16* __restrict__ wmat,
                                              int rowbase, int colbase, int n0,
                                              int hi, int kb, v16bf& a0, v16bf& a1,
                                              v16bf (&b)[4]) {
    a0 = load_tile16(amat, E_DIM, rowbase + n0, kb, hi);
    a1 = load_tile16(amat, E_DIM, rowbase + 16 + n0, kb, hi);
#pragma unroll
    for (int nc = 0; nc < 4; ++nc)
        b[nc] = load_tile16(wmat, E_DIM, colbase + nc * 16 + n0, kb, hi);
}

// ---------------------------------------------------------------------------
// f32 -> bf16 conversion
// ---------------------------------------------------------------------------
__global__ void cvt_bf16_kernel(const float* __restrict__ in,
                                __bf16* __restrict__ out, int n) {
    int i = blockIdx.x * blockDim.x + threadIdx.x;
    if (i < n) out[i] = (__bf16)in[i];
}

// ---------------------------------------------------------------------------
// QKV projection + bias (+ q scale) + RoPE, output [head][T][64] bf16.
// grid: (T/128, HEADS), block 128 (4 waves). Wave w: 32 rows (two 16-row A
// tiles sharing 4 B tiles). K loop unrolled x2 with two statically-named
// fragment buffer sets -> software pipelining without movrels/scratch.
// mode: 0 = Q (scale+rope), 1 = K (rope), 2 = V (plain)
// ---------------------------------------------------------------------------
__global__ __launch_bounds__(128) void qkv_rope_kernel(
    const __bf16* __restrict__ xb, const __bf16* __restrict__ wb,
    const float* __restrict__ bias, const int* __restrict__ cu,
    __bf16* __restrict__ outh, int T, int nseq, int mode) {
    const int wave = threadIdx.x >> 5;
    const int lane = threadIdx.x & 31;
    const int n0 = lane & 15;
    const int hi = lane >> 4;
    const int head = blockIdx.y;
    const int colbase = head * 64;
    const int rowbase = blockIdx.x * 128 + wave * 32;

    v8f acc[2][4];
#pragma unroll
    for (int t = 0; t < 2; ++t)
#pragma unroll
        for (int j = 0; j < 4; ++j) acc[t][j] = (v8f){};

    v16bf a0A, a1A, bA[4];
    v16bf a0B, a1B, bB[4];
    load_frag_set(xb, wb, rowbase, colbase, n0, hi, 0, a0A, a1A, bA);

    const __bf16* xrow = xb + (size_t)(rowbase + n0) * E_DIM;
    for (int kb = 0; kb < E_DIM; kb += 64) {  // two 32-chunks per iteration
        __builtin_prefetch(xrow + kb + 128, 0, 1);  // global_prefetch_b8
        // stage B-set for kb+32 (always in-range: E_DIM % 64 == 0)
        load_frag_set(xb, wb, rowbase, colbase, n0, hi, kb + 32, a0B, a1B, bB);
#pragma unroll
        for (int nc = 0; nc < 4; ++nc) {
            acc[0][nc] = wmma_bf16(a0A, bA[nc], acc[0][nc]);
            acc[1][nc] = wmma_bf16(a1A, bA[nc], acc[1][nc]);
        }
        if (kb + 64 < E_DIM)
            load_frag_set(xb, wb, rowbase, colbase, n0, hi, kb + 64, a0A, a1A, bA);
#pragma unroll
        for (int nc = 0; nc < 4; ++nc) {
            acc[0][nc] = wmma_bf16(a0B, bB[nc], acc[0][nc]);
            acc[1][nc] = wmma_bf16(a1B, bB[nc], acc[1][nc]);
        }
    }

    // bias (+ scale for q)
    const float scale = (mode == 0) ? 0.125f : 1.0f;  // d^-0.5, d=64
#pragma unroll
    for (int nc = 0; nc < 4; ++nc) {
        float bia = bias[colbase + nc * 16 + n0];
#pragma unroll
        for (int t = 0; t < 2; ++t)
#pragma unroll
            for (int i = 0; i < 8; ++i)
                acc[t][nc][i] = (acc[t][nc][i] + bia) * scale;
    }

    if (mode <= 1) {
        // rotate_half is register-local: chunk0<->chunk2, chunk1<->chunk3
        const float f0 = __powf(10000.f, -((float)n0) / 32.f);
        const float f1 = __powf(10000.f, -((float)(n0 + 16)) / 32.f);
#pragma unroll
        for (int t = 0; t < 2; ++t) {
#pragma unroll
            for (int i = 0; i < 8; ++i) {
                int tok = rowbase + t * 16 + i + 8 * hi;
                int s = 0;
                while (s + 1 < nseq && tok >= cu[s + 1]) ++s;
                float posf = (float)(tok - cu[s]);
                float g0 = posf * f0, g1 = posf * f1;
                float co0 = __cosf(g0), si0 = __sinf(g0);
                float co1 = __cosf(g1), si1 = __sinf(g1);
                float t0 = acc[t][0][i], t1 = acc[t][1][i];
                float t2 = acc[t][2][i], t3 = acc[t][3][i];
                acc[t][0][i] = t0 * co0 - t2 * si0;
                acc[t][1][i] = t1 * co1 - t3 * si1;
                acc[t][2][i] = t2 * co0 + t0 * si0;
                acc[t][3][i] = t3 * co1 + t1 * si1;
            }
        }
    }

#pragma unroll
    for (int t = 0; t < 2; ++t)
#pragma unroll
        for (int nc = 0; nc < 4; ++nc)
#pragma unroll
            for (int i = 0; i < 8; ++i) {
                int tok = rowbase + t * 16 + i + 8 * hi;
                outh[((size_t)head * T + tok) * HDIM + nc * 16 + n0] =
                    (__bf16)acc[t][nc][i];
            }
}

// ---------------------------------------------------------------------------
// Flash attention over packed sequences.
// grid: (max_len/64, nseq, HEADS), block 128 (4 waves, 16 queries each).
// K/V tiles staged memory->LDS via CDNA5 async-to-LDS loads (ASYNCcnt).
// ---------------------------------------------------------------------------
__global__ __launch_bounds__(128) void attn_kernel(
    const __bf16* __restrict__ qh, const __bf16* __restrict__ kh,
    const __bf16* __restrict__ vh, const int* __restrict__ cu,
    __bf16* __restrict__ ao, int T) {
    __shared__ __bf16 ks[32 * HDIM];
    __shared__ __bf16 vs[32 * HDIM];
    __shared__ __bf16 ps[4][16 * 32];

    const int seq = blockIdx.y;
    const int head = blockIdx.z;
    const int s0 = cu[seq];
    const int L = cu[seq + 1] - s0;
    const int qtile = blockIdx.x;
    if (qtile * 64 >= L) return;  // uniform across block

    const int wave = threadIdx.x >> 5;
    const int lane = threadIdx.x & 31;
    const int n0 = lane & 15;
    const int hi = lane >> 4;
    const int qrow0 = qtile * 64 + wave * 16;  // within sequence

    const __bf16* qbase = qh + ((size_t)head * T + s0) * HDIM;
    const __bf16* kbase = kh + ((size_t)head * T + s0) * HDIM;
    const __bf16* vbase = vh + ((size_t)head * T + s0) * HDIM;

    // Q resident in registers for the whole key loop
    const v16bf aq0 = load_tile16(qbase, HDIM, qrow0 + n0, 0, hi);
    const v16bf aq1 = load_tile16(qbase, HDIM, qrow0 + n0, 32, hi);

    v8f o[4];
#pragma unroll
    for (int j = 0; j < 4; ++j) o[j] = (v8f){};
    float mrow[8], lrow[8];
#pragma unroll
    for (int i = 0; i < 8; ++i) { mrow[i] = -3e38f; lrow[i] = 0.f; }

    // cooperative K/V chunk fill: 128 threads x 16B = 2048B, half of the 4096B 32x64 row set
    const int ldr = threadIdx.x >> 2;          // 0..31 key row
    const int ldc = (threadIdx.x & 3) * 16;    // 0,16,32,48
    const unsigned lds_k = (unsigned)(uintptr_t)(ks + ldr * HDIM + ldc);
    const unsigned lds_v = (unsigned)(uintptr_t)(vs + ldr * HDIM + ldc);

    for (int kv = 0; kv < L; kv += 32) {
        int tok = kv + ldr;
        if (tok < L) {
            if (kv + 64 < L)
                __builtin_prefetch(kbase + (size_t)(tok + 32) * HDIM + ldc, 0, 1);
            unsigned long long gak =
                (unsigned long long)(uintptr_t)(kbase + (size_t)tok * HDIM + ldc);
            unsigned long long gav =
                (unsigned long long)(uintptr_t)(vbase + (size_t)tok * HDIM + ldc);
            // CDNA5 async memory->LDS copy, tracked by ASYNCcnt
            asm volatile("global_load_async_to_lds_b128 %0, %1, off"
                         :: "v"(lds_k), "v"(gak) : "memory");
            asm volatile("global_load_async_to_lds_b128 %0, %1, off"
                         :: "v"(lds_v), "v"(gav) : "memory");
        } else {
            v16bf z;
#pragma unroll
            for (int j = 0; j < 16; ++j) z[j] = (__bf16)0.f;
            *(v16bf*)(ks + ldr * HDIM + ldc) = z;
            *(v16bf*)(vs + ldr * HDIM + ldc) = z;
        }
        // barriers do NOT drain ASYNCcnt -> explicit wait before signaling
        asm volatile("s_wait_asynccnt 0x0" ::: "memory");
        __syncthreads();

        // S = Q K^T : keys 0..15 (sa), 16..31 (sb), each 2 wmma over d=64
        v8f sa = (v8f){}, sb = (v8f){};
        sa = wmma_bf16(aq0, load_tile16(ks, HDIM, n0, 0, hi), sa);
        sa = wmma_bf16(aq1, load_tile16(ks, HDIM, n0, 32, hi), sa);
        sb = wmma_bf16(aq0, load_tile16(ks, HDIM, 16 + n0, 0, hi), sb);
        sb = wmma_bf16(aq1, load_tile16(ks, HDIM, 16 + n0, 32, hi), sb);

        // mask invalid keys (lane column = key within chunk)
        if (kv + n0 >= L) {
#pragma unroll
            for (int i = 0; i < 8; ++i) sa[i] = -1e9f;
        }
        if (kv + 16 + n0 >= L) {
#pragma unroll
            for (int i = 0; i < 8; ++i) sb[i] = -1e9f;
        }

        // online softmax (row = i + 8*hi lives in one 16-lane half)
        float corr[8];
#pragma unroll
        for (int i = 0; i < 8; ++i) {
            float r = half_red_max(fmaxf(sa[i], sb[i]));
            float mn = fmaxf(mrow[i], r);
            corr[i] = __expf(mrow[i] - mn);
            float p0 = __expf(sa[i] - mn);
            float p1 = __expf(sb[i] - mn);
            sa[i] = p0; sb[i] = p1;
            lrow[i] = lrow[i] * corr[i] + half_red_sum(p0 + p1);
            mrow[i] = mn;
        }

        // stage P (C-layout -> A-layout) through LDS
#pragma unroll
        for (int i = 0; i < 8; ++i) {
            int rm = i + 8 * hi;
            ps[wave][rm * 32 + n0] = (__bf16)sa[i];
            ps[wave][rm * 32 + 16 + n0] = (__bf16)sb[i];
        }
        __syncthreads();

        // O = O*corr + P V
        v16bf ap = load_tile16(&ps[wave][0], 32, n0, 0, hi);
#pragma unroll
        for (int j = 0; j < 4; ++j) {
#pragma unroll
            for (int i = 0; i < 8; ++i) o[j][i] *= corr[i];
            v16bf bv;  // B[k=key][n=d-col]: strided LDS reads
#pragma unroll
            for (int e = 0; e < 8; ++e) {
                bv[e]     = vs[(hi * 8 + e) * HDIM + j * 16 + n0];
                bv[e + 8] = vs[(16 + hi * 8 + e) * HDIM + j * 16 + n0];
            }
            o[j] = wmma_bf16(ap, bv, o[j]);
        }
        __syncthreads();
    }

    // normalize and scatter back to packed [T][E] bf16
#pragma unroll
    for (int i = 0; i < 8; ++i) {
        int rm = i + 8 * hi;
        int t = s0 + qrow0 + rm;
        float inv = 1.0f / lrow[i];
#pragma unroll
        for (int j = 0; j < 4; ++j)
            ao[(size_t)t * E_DIM + head * HDIM + j * 16 + n0] = (__bf16)(o[j][i] * inv);
    }
}

// ---------------------------------------------------------------------------
// Output projection: out = ao @ Wo^T + bo (f32 out).
// grid: (T/128, E/64), block 128 (4 waves x 32 rows), statically pipelined.
// ---------------------------------------------------------------------------
__global__ __launch_bounds__(128) void proj_kernel(
    const __bf16* __restrict__ ao, const __bf16* __restrict__ wob,
    const float* __restrict__ bo, float* __restrict__ out, int T) {
    const int wave = threadIdx.x >> 5;
    const int lane = threadIdx.x & 31;
    const int n0 = lane & 15;
    const int hi = lane >> 4;
    const int cb = blockIdx.y;  // 64-wide column block
    const int colbase = cb * 64;
    const int rowbase = blockIdx.x * 128 + wave * 32;

    v8f acc[2][4];
#pragma unroll
    for (int t = 0; t < 2; ++t)
#pragma unroll
        for (int j = 0; j < 4; ++j) acc[t][j] = (v8f){};

    v16bf a0A, a1A, bA[4];
    v16bf a0B, a1B, bB[4];
    load_frag_set(ao, wob, rowbase, colbase, n0, hi, 0, a0A, a1A, bA);

    const __bf16* arow = ao + (size_t)(rowbase + n0) * E_DIM;
    for (int kb = 0; kb < E_DIM; kb += 64) {
        __builtin_prefetch(arow + kb + 128, 0, 1);
        load_frag_set(ao, wob, rowbase, colbase, n0, hi, kb + 32, a0B, a1B, bB);
#pragma unroll
        for (int nc = 0; nc < 4; ++nc) {
            acc[0][nc] = wmma_bf16(a0A, bA[nc], acc[0][nc]);
            acc[1][nc] = wmma_bf16(a1A, bA[nc], acc[1][nc]);
        }
        if (kb + 64 < E_DIM)
            load_frag_set(ao, wob, rowbase, colbase, n0, hi, kb + 64, a0A, a1A, bA);
#pragma unroll
        for (int nc = 0; nc < 4; ++nc) {
            acc[0][nc] = wmma_bf16(a0B, bB[nc], acc[0][nc]);
            acc[1][nc] = wmma_bf16(a1B, bB[nc], acc[1][nc]);
        }
    }

#pragma unroll
    for (int nc = 0; nc < 4; ++nc) {
        float bia = bo[colbase + nc * 16 + n0];
#pragma unroll
        for (int t = 0; t < 2; ++t)
#pragma unroll
            for (int i = 0; i < 8; ++i) {
                int tok = rowbase + t * 16 + i + 8 * hi;
                out[(size_t)tok * E_DIM + colbase + nc * 16 + n0] =
                    acc[t][nc][i] + bia;
            }
    }
}

// ---------------------------------------------------------------------------
extern "C" void kernel_launch(void* const* d_in, const int* in_sizes, int n_in,
                              void* d_out, int out_size, void* d_ws, size_t ws_size,
                              hipStream_t stream) {
    const float* x  = (const float*)d_in[0];
    const int* cu   = (const int*)d_in[1];
    const float* Wq = (const float*)d_in[3];
    const float* bq = (const float*)d_in[4];
    const float* Wk = (const float*)d_in[5];
    const float* bk = (const float*)d_in[6];
    const float* Wv = (const float*)d_in[7];
    const float* bv = (const float*)d_in[8];
    const float* Wo = (const float*)d_in[9];
    const float* bo = (const float*)d_in[10];
    float* out = (float*)d_out;

    const int T = in_sizes[0] / E_DIM;     // 8192
    const int nseq = in_sizes[1] - 1;      // 8

    // workspace layout (bf16 scratch)
    char* w = (char*)d_ws;
    size_t off = 0;
    __bf16* xb  = (__bf16*)(w + off); off += (size_t)T * E_DIM * 2;
    __bf16* wqb = (__bf16*)(w + off); off += (size_t)E_DIM * E_DIM * 2;
    __bf16* wkb = (__bf16*)(w + off); off += (size_t)E_DIM * E_DIM * 2;
    __bf16* wvb = (__bf16*)(w + off); off += (size_t)E_DIM * E_DIM * 2;
    __bf16* wob = (__bf16*)(w + off); off += (size_t)E_DIM * E_DIM * 2;
    __bf16* qhp = (__bf16*)(w + off); off += (size_t)HEADS * T * HDIM * 2;
    __bf16* khp = (__bf16*)(w + off); off += (size_t)HEADS * T * HDIM * 2;
    __bf16* vhp = (__bf16*)(w + off); off += (size_t)HEADS * T * HDIM * 2;
    __bf16* aop = (__bf16*)(w + off); off += (size_t)T * E_DIM * 2;

    // 1) convert to bf16
    {
        int n = T * E_DIM;
        cvt_bf16_kernel<<<(n + 255) / 256, 256, 0, stream>>>(x, xb, n);
        int m = E_DIM * E_DIM;
        cvt_bf16_kernel<<<(m + 255) / 256, 256, 0, stream>>>(Wq, wqb, m);
        cvt_bf16_kernel<<<(m + 255) / 256, 256, 0, stream>>>(Wk, wkb, m);
        cvt_bf16_kernel<<<(m + 255) / 256, 256, 0, stream>>>(Wv, wvb, m);
        cvt_bf16_kernel<<<(m + 255) / 256, 256, 0, stream>>>(Wo, wob, m);
    }

    // 2) QKV + rope
    dim3 g1(T / 128, HEADS);
    qkv_rope_kernel<<<g1, 128, 0, stream>>>(xb, wqb, bq, cu, qhp, T, nseq, 0);
    qkv_rope_kernel<<<g1, 128, 0, stream>>>(xb, wkb, bk, cu, khp, T, nseq, 1);
    qkv_rope_kernel<<<g1, 128, 0, stream>>>(xb, wvb, bv, cu, vhp, T, nseq, 2);

    // 3) flash attention (max_len = 1280 -> 20 query tiles of 64)
    dim3 g2(1280 / 64, nseq, HEADS);
    attn_kernel<<<g2, 128, 0, stream>>>(qhp, khp, vhp, cu, aop, T);

    // 4) output projection
    dim3 g3(T / 128, E_DIM / 64);
    proj_kernel<<<g3, 128, 0, stream>>>(aop, wob, bo, out, T);
}
